// ToMeSelfAttention_64390149702059
// MI455X (gfx1250) — matrix operand order repaired and checked by reading, not compiled, not run on hardware
//
#include <hip/hip_runtime.h>
#include <stdint.h>

// ---------------------------------------------------------------------------
// ToMe proportional self-attention forward for MI455X (gfx1250, wave32, WMMA)
//   B=8, N=1024, C=768, H=12, HD=64
// GEMMs: v_wmma_f32_16x16x32_bf16. Shared B-operands staged to LDS with
// global_load_async_to_lds_b128 (double-buffered, ASYNCcnt-synchronized).
// ---------------------------------------------------------------------------

#define B_  8
#define N_  1024
#define H_  12
#define HD_ 64
#define C_  768

typedef __bf16 bf16_t;
typedef __attribute__((ext_vector_type(16))) __bf16 bf16x16;
typedef __attribute__((ext_vector_type(8)))  __bf16 bf16x8;
typedef __attribute__((ext_vector_type(4)))  __bf16 bf16x4;
typedef __attribute__((ext_vector_type(8)))  float  floatx8;
typedef __attribute__((ext_vector_type(4)))  float  floatx4;

__device__ __forceinline__ floatx8 wmma_bf16(bf16x16 a, bf16x16 b, floatx8 c) {
  return __builtin_amdgcn_wmma_f32_16x16x32_bf16(false, a, false, b,
                                                 (short)0, c, false, false);
}

// --- CDNA5 async DMA: global -> LDS, 16B per lane, tracked by ASYNCcnt ---
__device__ __forceinline__ void async_b128(uint32_t lds_off, uint64_t gbase,
                                           uint32_t goff) {
  asm volatile("global_load_async_to_lds_b128 %0, %1, %2 offset:0"
               :: "v"(lds_off), "v"(goff), "s"(gbase) : "memory");
}
__device__ __forceinline__ void wait_async0() {
  asm volatile("s_wait_asynccnt 0" ::: "memory");
}
__device__ __forceinline__ uint32_t lds_off(const void* p) {
  return (uint32_t)(uintptr_t)p;   // generic LDS addr: low 32 bits = LDS offset
}

// A fragment (16x32 M-major): lane L / L+16 hold row (L&15);
// elems 0..7 = K[half*8 .. +7], elems 8..15 = K[16+half*8 .. +7].
__device__ __forceinline__ bf16x16 afrag_bf16(const bf16_t* rowp, int k0, int half) {
  bf16x8 lo = *(const bf16x8*)(rowp + k0 + half * 8);
  bf16x8 hi = *(const bf16x8*)(rowp + k0 + 16 + half * 8);
  bf16x16 r;
#pragma unroll
  for (int i = 0; i < 8; ++i) { r[i] = lo[i]; r[8 + i] = hi[i]; }
  return r;
}

// B fragment (32x16 KxN): lane holds col (lane&15); elems 0..7 =
// K[k0+half*8 .. +7], elems 8..15 = K[k0+16+half*8 .. +7] (rowp = that col's K-row).
__device__ __forceinline__ bf16x16 bfrag_bf16(const bf16_t* rowp, int k0, int half) {
  bf16x8 lo = *(const bf16x8*)(rowp + k0 + half * 8);
  bf16x8 hi = *(const bf16x8*)(rowp + k0 + 16 + half * 8);
  bf16x16 r;
#pragma unroll
  for (int i = 0; i < 8; ++i) { r[i] = lo[i]; r[8 + i] = hi[i]; }
  return r;
}

// Reductions across the 16-lane half sharing one C/D-layout row.
__device__ __forceinline__ float hmax16(float v) {
#pragma unroll
  for (int m = 1; m < 16; m <<= 1) v = fmaxf(v, __shfl_xor(v, m, 32));
  return v;
}
__device__ __forceinline__ float hsum16(float v) {
#pragma unroll
  for (int m = 1; m < 16; m <<= 1) v += __shfl_xor(v, m, 32);
  return v;
}

// ---------------------------------------------------------------------------
// Kernel 0: f32 -> bf16 bulk convert (removes cvt VALU from GEMM hot loops)
// ---------------------------------------------------------------------------
__global__ void cvt_f32_bf16_kernel(const float* __restrict__ src,
                                    bf16_t* __restrict__ dst, int n4) {
  const int i = blockIdx.x * blockDim.x + threadIdx.x;
  if (i >= n4) return;
  floatx4 v = *(const floatx4*)(src + (size_t)i * 4);
  bf16x4 o;
#pragma unroll
  for (int j = 0; j < 4; ++j) o[j] = (__bf16)v[j];
  *(bf16x4*)(dst + (size_t)i * 4) = o;
}

// ---------------------------------------------------------------------------
// Kernel 1: QKV projection  Xbf[8192,768] @ Wbf^T -> Q(*0.125)/K/V^T (bf16)
// grid (128, 36), block 128 (4 waves x 16 rows). Weight tile async-staged.
// ---------------------------------------------------------------------------
__launch_bounds__(128)
__global__ void qkv_wmma_kernel(const bf16_t* __restrict__ X,
                                const bf16_t* __restrict__ W,
                                bf16_t* __restrict__ Q,
                                bf16_t* __restrict__ Ko,
                                bf16_t* __restrict__ Vt) {
  __shared__ bf16_t btile[2][64 * 32];   // [col][k] double-buffered (8KB)

  const int tid = threadIdx.x;
  const int wave = tid >> 5, lane = tid & 31;
  const int half = lane >> 4, l15 = lane & 15;
  const int mt = blockIdx.x, nt = blockIdx.y;
  const int rowbase = mt * 64 + wave * 16;

  const bf16_t* xrow = X + (size_t)(rowbase + l15) * C_;
  const uint64_t wbase = (uint64_t)(uintptr_t)(W + (size_t)nt * 64 * C_);

  // prologue: stage k0=0 into buffer 0 (2 x b128 per lane = 4KB/WG)
#pragma unroll
  for (int j = 0; j < 2; ++j) {
    const int e = j * 128 + tid, col = e >> 2, c = e & 3;
    async_b128(lds_off(&btile[0][col * 32 + c * 8]), wbase,
               (uint32_t)(((uint32_t)col * C_ + c * 8) * 2));
  }

  floatx8 acc0 = {}, acc1 = {}, acc2 = {}, acc3 = {};
  for (int k0 = 0, it = 0; k0 < C_; k0 += 32, ++it) {
    wait_async0();
    __syncthreads();                       // current buffer visible to all
    const int cur = it & 1;
    if (k0 + 32 < C_) {
#pragma unroll
      for (int j = 0; j < 2; ++j) {
        const int e = j * 128 + tid, col = e >> 2, c = e & 3;
        async_b128(lds_off(&btile[cur ^ 1][col * 32 + c * 8]), wbase,
                   (uint32_t)(((uint32_t)col * C_ + (k0 + 32) + c * 8) * 2));
      }
    }
    bf16x16 a = afrag_bf16(xrow, k0, half);
    const bf16_t* bt = &btile[cur][0];
    acc0 = wmma_bf16(a, bfrag_bf16(bt + (0  + l15) * 32, 0, half), acc0);
    acc1 = wmma_bf16(a, bfrag_bf16(bt + (16 + l15) * 32, 0, half), acc1);
    acc2 = wmma_bf16(a, bfrag_bf16(bt + (32 + l15) * 32, 0, half), acc2);
    acc3 = wmma_bf16(a, bfrag_bf16(bt + (48 + l15) * 32, 0, half), acc3);
    __syncthreads();                       // done reading before overwrite
  }

  const int which = nt / H_;  // 0:q 1:k 2:v
  const int h = nt % H_;
#pragma unroll
  for (int r = 0; r < 8; ++r) {
    const int row = rowbase + r + 8 * half;
    const int b = row >> 10, n = row & (N_ - 1);
    const size_t hbase = ((size_t)(b * H_ + h) * N_ + n) * HD_;
    const float v0 = acc0[r], v1 = acc1[r], v2 = acc2[r], v3 = acc3[r];
    if (which == 0) {            // fold attention scale hd^-0.5 = 0.125
      Q[hbase + 0  + l15] = (bf16_t)(v0 * 0.125f);
      Q[hbase + 16 + l15] = (bf16_t)(v1 * 0.125f);
      Q[hbase + 32 + l15] = (bf16_t)(v2 * 0.125f);
      Q[hbase + 48 + l15] = (bf16_t)(v3 * 0.125f);
    } else if (which == 1) {
      Ko[hbase + 0  + l15] = (bf16_t)v0;
      Ko[hbase + 16 + l15] = (bf16_t)v1;
      Ko[hbase + 32 + l15] = (bf16_t)v2;
      Ko[hbase + 48 + l15] = (bf16_t)v3;
    } else {                     // V stored transposed: Vt[bh][d][n]
      const size_t vb = (size_t)(b * H_ + h) * HD_ * N_ + n;
      Vt[vb + (size_t)(0  + l15) * N_] = (bf16_t)v0;
      Vt[vb + (size_t)(16 + l15) * N_] = (bf16_t)v1;
      Vt[vb + (size_t)(32 + l15) * N_] = (bf16_t)v2;
      Vt[vb + (size_t)(48 + l15) * N_] = (bf16_t)v3;
    }
  }
}

// ---------------------------------------------------------------------------
// Kernel 2: flash attention, online softmax + log(size) bias + key mask.
// grid 1536 = B*H*16, block 128 (4 waves x 16 query rows).
// K/V^T tiles async-staged to LDS (shared by all 4 waves), double-buffered.
// ---------------------------------------------------------------------------
__launch_bounds__(128)
__global__ void attn_flash_kernel(const float* __restrict__ sizev,
                                  const int* __restrict__ kpm,
                                  const bf16_t* __restrict__ Q,
                                  const bf16_t* __restrict__ K,
                                  const bf16_t* __restrict__ Vt,
                                  bf16_t* __restrict__ O) {
  __shared__ bf16_t ktile[2][32 * 64];   // [token][d]   (8KB)
  __shared__ bf16_t vtile[2][64 * 32];   // [d][key]     (8KB)
  __shared__ bf16_t pstage[4][16 * 32];  // per-wave P transpose (4KB)

  const int tid = threadIdx.x;
  const int wave = tid >> 5, lane = tid & 31;
  const int half = lane >> 4, l15 = lane & 15;
  const int gx = blockIdx.x;
  const int rt = gx & 15;
  const int bh = gx >> 4;                 // b*H + h
  const int b = bh / H_, h = bh % H_;
  const int rowbase = rt * 64 + wave * 16;

  const size_t headQK = (size_t)bh * N_ * HD_;
  const uint64_t kbase = (uint64_t)(uintptr_t)(K + headQK);
  const uint64_t vbase = (uint64_t)(uintptr_t)(Vt + (size_t)bh * HD_ * N_);

  const bf16_t* qrow = Q + headQK + (size_t)(rowbase + l15) * HD_;
  const bf16x16 qa0 = afrag_bf16(qrow, 0,  half);
  const bf16x16 qa1 = afrag_bf16(qrow, 32, half);

  floatx8 o0 = {}, o1 = {}, o2 = {}, o3 = {};
  float mrun[8], lrun[8];
#pragma unroll
  for (int r = 0; r < 8; ++r) { mrun[r] = -3.0e30f; lrun[r] = 0.0f; }

  const float* szb = sizev + (size_t)b * N_;
  const int*   mb  = kpm   + (size_t)b * N_;
  bf16_t* st = &pstage[wave][0];

  // prologue: stage key tile 0 (K: 4KB contiguous; V^T: 64 rows x 64B)
#pragma unroll
  for (int j = 0; j < 2; ++j) {
    const int e = j * 128 + tid;
    async_b128(lds_off(&ktile[0][e * 8]), kbase, (uint32_t)(e * 16));
    const int d = e >> 2, c = e & 3;
    async_b128(lds_off(&vtile[0][d * 32 + c * 8]), vbase,
               (uint32_t)(((uint32_t)d * N_ + c * 8) * 2));
  }

  for (int it = 0; it < N_ / 32; ++it) {
    const int kt = it * 32;
    wait_async0();
    __syncthreads();
    const int cur = it & 1;
    if (kt + 32 < N_) {
      const int nxt = cur ^ 1;
#pragma unroll
      for (int j = 0; j < 2; ++j) {
        const int e = j * 128 + tid;
        async_b128(lds_off(&ktile[nxt][e * 8]), kbase,
                   (uint32_t)(((uint32_t)(kt + 32) * HD_ + e * 8) * 2));
        const int d = e >> 2, c = e & 3;
        async_b128(lds_off(&vtile[nxt][d * 32 + c * 8]), vbase,
                   (uint32_t)(((uint32_t)d * N_ + (kt + 32) + c * 8) * 2));
      }
    }

    // ---- S = (Q*scale) K^T for two 16-key column tiles (from LDS) ----
    const bf16_t* ktl = &ktile[cur][0];
    floatx8 s0 = {}, s1 = {};
    s0 = wmma_bf16(qa0, bfrag_bf16(ktl + (0  + l15) * 64, 0,  half), s0);
    s0 = wmma_bf16(qa1, bfrag_bf16(ktl + (0  + l15) * 64, 32, half), s0);
    s1 = wmma_bf16(qa0, bfrag_bf16(ktl + (16 + l15) * 64, 0,  half), s1);
    s1 = wmma_bf16(qa1, bfrag_bf16(ktl + (16 + l15) * 64, 32, half), s1);

    // ---- additive log-size bias / masked keys ----
    const int key0 = kt + l15, key1 = kt + 16 + l15;
    const float bias0 = mb[key0] ? -3.0e30f : __logf(szb[key0]);
    const float bias1 = mb[key1] ? -3.0e30f : __logf(szb[key1]);
#pragma unroll
    for (int r = 0; r < 8; ++r) { s0[r] += bias0; s1[r] += bias1; }

    // ---- online softmax (per-half row stats, vgpr-indexed) ----
    float cf[8];
#pragma unroll
    for (int r = 0; r < 8; ++r) {
      const float tmax = hmax16(fmaxf(s0[r], s1[r]));
      const float mnew = fmaxf(mrun[r], tmax);
      cf[r] = __expf(mrun[r] - mnew);
      mrun[r] = mnew;
      const float p0 = __expf(s0[r] - mnew);
      const float p1 = __expf(s1[r] - mnew);
      s0[r] = p0; s1[r] = p1;
      lrun[r] = lrun[r] * cf[r] + hsum16(p0 + p1);
    }
#pragma unroll
    for (int r = 0; r < 8; ++r) {
      o0[r] *= cf[r]; o1[r] *= cf[r]; o2[r] *= cf[r]; o3[r] *= cf[r];
    }

    // ---- stage P (D-layout) -> per-wave LDS row-major, read back as A ----
#pragma unroll
    for (int r = 0; r < 8; ++r) {
      st[(r + 8 * half) * 32 + l15]      = (bf16_t)s0[r];
      st[(r + 8 * half) * 32 + 16 + l15] = (bf16_t)s1[r];
    }
    asm volatile("s_wait_dscnt 0" ::: "memory");  // intra-wave LDS RAW
    const bf16x16 pa = afrag_bf16(st + l15 * 32, 0, half);

    // ---- O += P V  (V^T rows from LDS: contiguous local keys) ----
    const bf16_t* vtl = &vtile[cur][0];
    o0 = wmma_bf16(pa, bfrag_bf16(vtl + (0  + l15) * 32, 0, half), o0);
    o1 = wmma_bf16(pa, bfrag_bf16(vtl + (16 + l15) * 32, 0, half), o1);
    o2 = wmma_bf16(pa, bfrag_bf16(vtl + (32 + l15) * 32, 0, half), o2);
    o3 = wmma_bf16(pa, bfrag_bf16(vtl + (48 + l15) * 32, 0, half), o3);

    __syncthreads();   // all waves done with cur before it is re-staged
  }

  // ---- normalize and store bf16 into [B,N,C] at col h*64+d ----
#pragma unroll
  for (int r = 0; r < 8; ++r) {
    const float inv = 1.0f / lrun[r];
    const int row = rowbase + r + 8 * half;
    const size_t base = ((size_t)b * N_ + row) * C_ + h * HD_;
    O[base + 0  + l15] = (bf16_t)(o0[r] * inv);
    O[base + 16 + l15] = (bf16_t)(o1[r] * inv);
    O[base + 32 + l15] = (bf16_t)(o2[r] * inv);
    O[base + 48 + l15] = (bf16_t)(o3[r] * inv);
  }
}

// ---------------------------------------------------------------------------
// Kernel 3: output projection  Obf[8192,768] @ Wp^T + b -> out (f32)
// grid (128, 12), block 128. Weight tile async-staged like kernel 1.
// ---------------------------------------------------------------------------
__launch_bounds__(128)
__global__ void proj_wmma_kernel(const bf16_t* __restrict__ Ob,
                                 const bf16_t* __restrict__ Wp,
                                 const float* __restrict__ bias,
                                 float* __restrict__ out) {
  __shared__ bf16_t btile[2][64 * 32];

  const int tid = threadIdx.x;
  const int wave = tid >> 5, lane = tid & 31;
  const int half = lane >> 4, l15 = lane & 15;
  const int mt = blockIdx.x, nt = blockIdx.y;
  const int rowbase = mt * 64 + wave * 16;

  const bf16_t* orow = Ob + (size_t)(rowbase + l15) * C_;
  const uint64_t wbase = (uint64_t)(uintptr_t)(Wp + (size_t)nt * 64 * C_);

#pragma unroll
  for (int j = 0; j < 2; ++j) {
    const int e = j * 128 + tid, col = e >> 2, c = e & 3;
    async_b128(lds_off(&btile[0][col * 32 + c * 8]), wbase,
               (uint32_t)(((uint32_t)col * C_ + c * 8) * 2));
  }

  floatx8 acc0 = {}, acc1 = {}, acc2 = {}, acc3 = {};
  for (int k0 = 0, it = 0; k0 < C_; k0 += 32, ++it) {
    wait_async0();
    __syncthreads();
    const int cur = it & 1;
    if (k0 + 32 < C_) {
#pragma unroll
      for (int j = 0; j < 2; ++j) {
        const int e = j * 128 + tid, col = e >> 2, c = e & 3;
        async_b128(lds_off(&btile[cur ^ 1][col * 32 + c * 8]), wbase,
                   (uint32_t)(((uint32_t)col * C_ + (k0 + 32) + c * 8) * 2));
      }
    }
    bf16x16 a = afrag_bf16(orow, k0, half);
    const bf16_t* bt = &btile[cur][0];
    acc0 = wmma_bf16(a, bfrag_bf16(bt + (0  + l15) * 32, 0, half), acc0);
    acc1 = wmma_bf16(a, bfrag_bf16(bt + (16 + l15) * 32, 0, half), acc1);
    acc2 = wmma_bf16(a, bfrag_bf16(bt + (32 + l15) * 32, 0, half), acc2);
    acc3 = wmma_bf16(a, bfrag_bf16(bt + (48 + l15) * 32, 0, half), acc3);
    __syncthreads();
  }

  const int col0 = nt * 64;
  const float b0 = bias[col0 + 0  + l15];
  const float b1 = bias[col0 + 16 + l15];
  const float b2 = bias[col0 + 32 + l15];
  const float b3 = bias[col0 + 48 + l15];
#pragma unroll
  for (int r = 0; r < 8; ++r) {
    const int row = rowbase + r + 8 * half;
    float* op = out + (size_t)row * C_ + col0;
    op[0  + l15] = acc0[r] + b0;
    op[16 + l15] = acc1[r] + b1;
    op[32 + l15] = acc2[r] + b2;
    op[48 + l15] = acc3[r] + b3;
  }
}

// ---------------------------------------------------------------------------
// Kernel 4: k_mean[b,n,d] = mean over heads of K
// ---------------------------------------------------------------------------
__global__ void kmean_kernel(const bf16_t* __restrict__ Kbf,
                             float* __restrict__ km) {
  const size_t i = (size_t)blockIdx.x * blockDim.x + threadIdx.x;  // B*N*HD
  const int d = (int)(i & 63);
  const size_t bn = i >> 6;
  const int n = (int)(bn & (N_ - 1));
  const int b = (int)(bn >> 10);
  float s = 0.0f;
#pragma unroll
  for (int h = 0; h < H_; ++h)
    s += (float)Kbf[((size_t)(b * H_ + h) * N_ + n) * HD_ + d];
  km[i] = s * (1.0f / 12.0f);
}

// ---------------------------------------------------------------------------
extern "C" void kernel_launch(void* const* d_in, const int* in_sizes, int n_in,
                              void* d_out, int out_size, void* d_ws, size_t ws_size,
                              hipStream_t stream) {
  const float* x     = (const float*)d_in[0];   // [8,1024,768]
  const float* sizev = (const float*)d_in[1];   // [8,1024,1]
  const int*   kpm   = (const int*)d_in[2];     // [8,1024] bool
  const float* wqkv  = (const float*)d_in[3];   // [2304,768]
  const float* wproj = (const float*)d_in[4];   // [768,768]
  const float* bproj = (const float*)d_in[5];   // [768]

  float* out   = (float*)d_out;                 // [8,1024,768]
  float* kmean = out + (size_t)B_ * N_ * C_;    // [8,1024,64]

  const size_t NX  = (size_t)B_ * N_ * C_;      // 6,291,456
  const size_t NW1 = (size_t)3 * C_ * C_;       // 1,769,472
  const size_t NW2 = (size_t)C_ * C_;           //   589,824
  const size_t HE  = (size_t)B_ * H_ * N_ * HD_;// 6,291,456

  bf16_t* Xbf  = (bf16_t*)d_ws;
  bf16_t* W1bf = Xbf + NX;
  bf16_t* W2bf = W1bf + NW1;
  bf16_t* Qbf  = W2bf + NW2;
  bf16_t* Kbf  = Qbf + HE;
  bf16_t* Vtb  = Kbf + HE;
  bf16_t* Obf  = Vtb + HE;                      // total ~67.6 MB

  cvt_f32_bf16_kernel<<<(int)(NX  / 4 + 255) / 256, 256, 0, stream>>>(x, Xbf, (int)(NX / 4));
  cvt_f32_bf16_kernel<<<(int)(NW1 / 4 + 255) / 256, 256, 0, stream>>>(wqkv, W1bf, (int)(NW1 / 4));
  cvt_f32_bf16_kernel<<<(int)(NW2 / 4 + 255) / 256, 256, 0, stream>>>(wproj, W2bf, (int)(NW2 / 4));

  qkv_wmma_kernel<<<dim3(128, 36), 128, 0, stream>>>(Xbf, W1bf, Qbf, Kbf, Vtb);
  attn_flash_kernel<<<B_ * H_ * 16, 128, 0, stream>>>(sizev, kpm, Qbf, Kbf, Vtb, Obf);
  proj_wmma_kernel<<<dim3(128, 12), 128, 0, stream>>>(Obf, wproj ? W2bf : W2bf, bproj, out);
  kmean_kernel<<<(B_ * N_ * HD_) / 256, 256, 0, stream>>>(Kbf, kmean);
}
